// IMCPMicroTransformer_2980707303896
// MI455X (gfx1250) — matrix operand order repaired and checked by reading, not compiled, not run on hardware
//
#include <hip/hip_runtime.h>
#include <hip/hip_bf16.h>

// ---- CDNA5 WMMA types --------------------------------------------------
typedef __attribute__((ext_vector_type(16))) __bf16        v16bf;
typedef __attribute__((ext_vector_type(8)))  float         v8f;
typedef __attribute__((ext_vector_type(4)))  unsigned int  uv4;
typedef __attribute__((ext_vector_type(8)))  unsigned int  uv8;
// ---- TDM descriptor vector types (6-arg builtin on this toolchain) -----
typedef __attribute__((ext_vector_type(4))) unsigned int tdm_g0_t;
typedef __attribute__((ext_vector_type(8))) int          tdm_g1_t;
typedef __attribute__((ext_vector_type(4))) int          tdm_g2_t;
typedef __attribute__((ext_vector_type(8))) int          tdm_g4_t;

__device__ __forceinline__ v8f wmma_bf16(v16bf a, v16bf b, v8f c) {
  // emits v_wmma_f32_16x16x32_bf16
  return __builtin_amdgcn_wmma_f32_16x16x32_bf16(false, a, false, b, (short)0, c,
                                                 false, false);
}

__device__ __forceinline__ v16bf combine32(uv4 a, uv4 b) {
  uv8 u;
  u[0] = a[0]; u[1] = a[1]; u[2] = a[2]; u[3] = a[3];
  u[4] = b[0]; u[5] = b[1]; u[6] = b[2]; u[7] = b[3];
  return __builtin_bit_cast(v16bf, u);
}

// A fragment (16x32 bf16, M x K): lanes 0-15 hold rows m, K chunks {0..7,16..23};
// lanes 16-31 hold K chunks {8..15,24..31}.  Row-major source, lda in elems.
__device__ __forceinline__ v16bf load_A(const __bf16* base, int lda, int mbase, int k0) {
  const int lane = threadIdx.x & 31;
  const int m  = mbase + (lane & 15);
  const int kk = k0 + ((lane >> 4) << 3);
  uv4 q0 = *(const uv4*)(base + m * lda + kk);
  uv4 q1 = *(const uv4*)(base + m * lda + kk + 16);
  return combine32(q0, q1);
}

// B fragment (32x16 bf16, K x N) from row-major W[n][k]: lane holds column
// n = lane&15, K chunks {0..7,16..23} (lanes 0-15) / {8..15,24..31} (lanes 16-31).
__device__ __forceinline__ v16bf load_B(const __bf16* base, int ldb, int nbase, int k0) {
  const int lane = threadIdx.x & 31;
  const int n  = nbase + (lane & 15);
  const int kk = k0 + ((lane >> 4) << 3);
  uv4 q0 = *(const uv4*)(base + n * ldb + kk);
  uv4 q1 = *(const uv4*)(base + n * ldb + kk + 16);
  return combine32(q0, q1);
}

// C layout: VGPR r, lanes 0-15 -> M = mbase+r, lanes 16-31 -> M = mbase+8+r; N = lane&15.
__device__ __forceinline__ void store_C(__bf16* dst, int ldc, int mbase, int nbase, v8f acc) {
  const int lane = threadIdx.x & 31;
  const int n  = nbase + (lane & 15);
  const int mb = mbase + ((lane >> 4) << 3);
#pragma unroll
  for (int r = 0; r < 8; ++r) dst[(mb + r) * ldc + n] = (__bf16)acc[r];
}

__device__ __forceinline__ void store_C_T(__bf16* dst, int ldc, int mbase, int nbase, v8f acc) {
  const int lane = threadIdx.x & 31;
  const int n  = nbase + (lane & 15);
  const int mb = mbase + ((lane >> 4) << 3);
#pragma unroll
  for (int r = 0; r < 8; ++r) dst[n * ldc + (mb + r)] = (__bf16)acc[r];
}

// reductions across 16-lane halves (matches C-matrix row spread)
__device__ __forceinline__ float redmax16(float x) {
#pragma unroll
  for (int off = 1; off < 16; off <<= 1) x = fmaxf(x, __shfl_xor(x, off, 32));
  return x;
}
__device__ __forceinline__ float redsum16(float x) {
#pragma unroll
  for (int off = 1; off < 16; off <<= 1) x += __shfl_xor(x, off, 32);
  return x;
}

// ---- TDM: DMA a 2D row-major tile (rows x cols bf16) from global into LDS ----
// D# built per CDNA5 ISA 8.3/8.4: group0 = {count=1, lds_addr, global_addr, type=2},
// group1 = {data_size=2B, tensor dims, tile dims, dim0 stride}; groups 2/3/4 zero (2-D).
__device__ __forceinline__ void tdm_load_2d(const void* gsrc, unsigned lds_off,
                                            unsigned rows, unsigned cols,
                                            unsigned stride_elems) {
  const unsigned long long ga = (unsigned long long)gsrc;
  tdm_g0_t g0;
  g0[0] = 1u;                                               // count=1, user mode
  g0[1] = lds_off;                                          // lds_addr (bytes)
  g0[2] = (unsigned)(ga & 0xFFFFFFFFu);                     // global_addr[31:0]
  g0[3] = (unsigned)((ga >> 32) & 0x01FFFFFFu) | (2u << 30);// addr[56:32] | type=2
  tdm_g1_t g1;
  g1[0] = (int)(1u << 16);                                  // data_size=1 (2 bytes)
  g1[1] = (int)(cols << 16);                                // tensor_dim0[15:0]
  g1[2] = (int)((cols >> 16) | (rows << 16));               // dim0 hi | tensor_dim1 lo
  g1[3] = (int)((rows >> 16) | (cols << 16));               // dim1 hi | tile_dim0
  g1[4] = (int)rows;                                        // tile_dim1 (tile_dim2=0)
  g1[5] = (int)stride_elems;                                // tensor_dim0_stride[31:0]
  g1[6] = 0;
  g1[7] = 0;
  tdm_g2_t z4 = {0, 0, 0, 0};
  tdm_g4_t z8 = {0, 0, 0, 0, 0, 0, 0, 0};
  __builtin_amdgcn_tensor_load_to_lds(g0, g1, z4, z4, z8, 0);
}

// ---- weight f32 -> bf16 pre-pack (runs once per launch, ~70K elems) ----
// ws layout (bf16 elems): inW[2*192*64] | outW[2*64*64] | ff1[2*128*64] |
//                         ff2[2*64*128] | head padded to 64x64 (rows>=50 zero)
#define WS_INW   0
#define WS_OUTW  24576
#define WS_FF1   32768
#define WS_FF2   49152
#define WS_HEAD  65536
#define WS_TOTAL 69632

__global__ __launch_bounds__(256) void cvtw(const float* __restrict__ in_w,
                                            const float* __restrict__ out_w,
                                            const float* __restrict__ ff1_w,
                                            const float* __restrict__ ff2_w,
                                            const float* __restrict__ head_w,
                                            __bf16* __restrict__ ws) {
  int i = blockIdx.x * 256 + threadIdx.x;
  if (i >= WS_TOTAL) return;
  float v;
  if      (i < WS_OUTW) v = in_w[i];
  else if (i < WS_FF1)  v = out_w[i - WS_OUTW];
  else if (i < WS_FF2)  v = ff1_w[i - WS_FF1];
  else if (i < WS_HEAD) v = ff2_w[i - WS_FF2];
  else { int j = i - WS_HEAD; v = ((j >> 6) < 50) ? head_w[j] : 0.0f; }
  ws[i] = (__bf16)v;
}

// LDS weight-staging layout (bf16 elems within wsm[])
#define L_INW  0        // 192x64
#define L_OUTW 12288    // 64x64
#define L_FF1  16384    // 128x64
#define L_FF2  24576    // 64x128
#define L_WTOT 32768    // 64 KB

// ---- fused transformer: one workgroup per sequence, 4 waves, 16 rows each ----
__global__ __launch_bounds__(128) void tfwd(
    const int* __restrict__ x, const unsigned char* __restrict__ pmask,
    const float* __restrict__ emb, const float* __restrict__ pos,
    const float* __restrict__ in_b, const float* __restrict__ out_b,
    const float* __restrict__ ln1_s, const float* __restrict__ ln1_b,
    const float* __restrict__ ff1_b, const float* __restrict__ ff2_b,
    const float* __restrict__ ln2_s, const float* __restrict__ ln2_b,
    const float* __restrict__ head_b,
    const __bf16* __restrict__ wIn, const __bf16* __restrict__ wOut,
    const __bf16* __restrict__ wFF1, const __bf16* __restrict__ wFF2,
    const __bf16* __restrict__ wHead, float* __restrict__ out) {
  __shared__ __bf16 hb[64 * 64];     // activations (residual stream), bf16
  __shared__ __bf16 bufA[64 * 128];  // Q|K during attention; FFN hidden later
  __shared__ __bf16 vt[64 * 64];     // V transposed: vt[d][t]
  __shared__ __bf16 probs[64 * 64];  // attention probabilities (per head)
  __shared__ __bf16 ctx[64 * 64];    // attention context
  __shared__ __bf16 wsm[L_WTOT];     // TDM-staged per-layer weights (64 KB)
  __shared__ int    toks[64];

  const int seq   = blockIdx.x;
  const int tid   = threadIdx.x;
  const int lane  = tid & 31;
  const int mbase = (tid >> 5) << 4;  // wave-owned 16-row strip

  if (tid == 0) { __builtin_prefetch(wHead, 0, 1); }

  if (tid < 64) toks[tid] = x[(size_t)seq * 64 + tid];
  __syncthreads();
  for (int i = tid; i < 64 * 64; i += 128) {
    const int t = i >> 6;
    hb[i] = (__bf16)(emb[toks[t] * 64 + (i & 63)] + pos[i]);
  }

  for (int l = 0; l < 2; ++l) {
    // ---- stage this layer's weights into LDS via the Tensor Data Mover ----
    if (tid < 32) {  // one wave issues the DMAs (EXEC ignored by TDM)
      const unsigned base = (unsigned)(unsigned long long)(const void*)wsm;
      tdm_load_2d(wIn  + l * 12288, base + 2 * L_INW , 192,  64,  64);
      tdm_load_2d(wOut + l * 4096 , base + 2 * L_OUTW,  64,  64,  64);
      tdm_load_2d(wFF1 + l * 8192 , base + 2 * L_FF1 , 128,  64,  64);
      tdm_load_2d(wFF2 + l * 8192 , base + 2 * L_FF2 ,  64, 128, 128);
      __builtin_amdgcn_s_wait_tensorcnt(0);
    }
    __syncthreads();

    // ---- QKV projection: [64x64] @ [64x192]^T, V stored transposed ----
    for (int nt = 0; nt < 12; ++nt) {
      const int nb = nt * 16;
      v8f acc = {};
#pragma unroll
      for (int ks = 0; ks < 2; ++ks)
        acc = wmma_bf16(load_A(hb, 64, mbase, ks * 32),
                        load_B(wsm + L_INW, 64, nb, ks * 32), acc);
      const float bias = in_b[l * 192 + nb + (lane & 15)];
#pragma unroll
      for (int r = 0; r < 8; ++r) acc[r] += bias;
      if (nt < 8) store_C(bufA, 128, mbase, nb, acc);          // Q cols 0..63, K cols 64..127
      else        store_C_T(vt, 64, mbase, nb - 128, acc);     // V^T
    }
    __syncthreads();

    // ---- attention per head (Dh=32 -> one WMMA per score tile) ----
    for (int hh = 0; hh < 2; ++hh) {
      v8f sc[4];
#pragma unroll
      for (int nt = 0; nt < 4; ++nt) {
        v8f z = {};
        sc[nt] = wmma_bf16(load_A(bufA, 128, mbase, hh * 32),
                           load_B(bufA + 64 + hh * 32, 128, nt * 16, 0), z);
        const float madd = pmask[(size_t)seq * 64 + nt * 16 + (lane & 15)] ? -1e9f : 0.0f;
#pragma unroll
        for (int r = 0; r < 8; ++r)
          sc[nt][r] = sc[nt][r] * 0.17677669529663687f + madd;  // 1/sqrt(32)
      }
#pragma unroll
      for (int r = 0; r < 8; ++r) {
        float mx = fmaxf(fmaxf(sc[0][r], sc[1][r]), fmaxf(sc[2][r], sc[3][r]));
        mx = redmax16(mx);
        float s = 0.0f;
#pragma unroll
        for (int nt = 0; nt < 4; ++nt) { sc[nt][r] = __expf(sc[nt][r] - mx); s += sc[nt][r]; }
        s = redsum16(s);
        const float inv = __frcp_rn(s);
#pragma unroll
        for (int nt = 0; nt < 4; ++nt) sc[nt][r] *= inv;
      }
#pragma unroll
      for (int nt = 0; nt < 4; ++nt) store_C(probs, 64, mbase, nt * 16, sc[nt]);
      __syncthreads();
      // ctx = probs @ V  (B operand contiguous thanks to vt)
#pragma unroll
      for (int nt = 0; nt < 2; ++nt) {
        const int nb = hh * 32 + nt * 16;
        v8f acc = {};
#pragma unroll
        for (int ks = 0; ks < 2; ++ks)
          acc = wmma_bf16(load_A(probs, 64, mbase, ks * 32), load_B(vt, 64, nb, ks * 32), acc);
        store_C(ctx, 64, mbase, nb, acc);
      }
      __syncthreads();
    }

    // ---- output projection + residual ----
    for (int nt = 0; nt < 4; ++nt) {
      const int nb = nt * 16;
      v8f acc = {};
#pragma unroll
      for (int ks = 0; ks < 2; ++ks)
        acc = wmma_bf16(load_A(ctx, 64, mbase, ks * 32),
                        load_B(wsm + L_OUTW, 64, nb, ks * 32), acc);
      const int n = nb + (lane & 15);
      const float bias = out_b[l * 64 + n];
      const int mb = mbase + ((lane >> 4) << 3);
#pragma unroll
      for (int r = 0; r < 8; ++r) {
        const int idx = (mb + r) * 64 + n;
        hb[idx] = (__bf16)((float)hb[idx] + acc[r] + bias);
      }
    }
    __syncthreads();
    // ---- LayerNorm 1 (one thread per token) ----
    if (tid < 64) {
      float sum = 0.0f, sq = 0.0f;
      for (int d = 0; d < 64; ++d) { const float v = (float)hb[tid * 64 + d]; sum += v; sq += v * v; }
      const float mean = sum * 0.015625f;
      const float var  = sq * 0.015625f - mean * mean;
      const float inv  = __frsqrt_rn(var + 1e-5f);
      for (int d = 0; d < 64; ++d) {
        const float y = ((float)hb[tid * 64 + d] - mean) * inv * ln1_s[l * 64 + d] + ln1_b[l * 64 + d];
        hb[tid * 64 + d] = (__bf16)y;
      }
    }
    __syncthreads();

    // ---- FFN: relu(h @ W1^T + b1) @ W2^T + b2, residual ----
    for (int nt = 0; nt < 8; ++nt) {
      const int nb = nt * 16;
      v8f acc = {};
#pragma unroll
      for (int ks = 0; ks < 2; ++ks)
        acc = wmma_bf16(load_A(hb, 64, mbase, ks * 32),
                        load_B(wsm + L_FF1, 64, nb, ks * 32), acc);
      const float bias = ff1_b[l * 128 + nb + (lane & 15)];
#pragma unroll
      for (int r = 0; r < 8; ++r) acc[r] = fmaxf(acc[r] + bias, 0.0f);
      store_C(bufA, 128, mbase, nb, acc);
    }
    __syncthreads();
    for (int nt = 0; nt < 4; ++nt) {
      const int nb = nt * 16;
      v8f acc = {};
#pragma unroll
      for (int ks = 0; ks < 4; ++ks)
        acc = wmma_bf16(load_A(bufA, 128, mbase, ks * 32),
                        load_B(wsm + L_FF2, 128, nb, ks * 32), acc);
      const int n = nb + (lane & 15);
      const float bias = ff2_b[l * 64 + n];
      const int mb = mbase + ((lane >> 4) << 3);
#pragma unroll
      for (int r = 0; r < 8; ++r) {
        const int idx = (mb + r) * 64 + n;
        hb[idx] = (__bf16)((float)hb[idx] + acc[r] + bias);
      }
    }
    __syncthreads();
    // ---- LayerNorm 2 ----
    if (tid < 64) {
      float sum = 0.0f, sq = 0.0f;
      for (int d = 0; d < 64; ++d) { const float v = (float)hb[tid * 64 + d]; sum += v; sq += v * v; }
      const float mean = sum * 0.015625f;
      const float var  = sq * 0.015625f - mean * mean;
      const float inv  = __frsqrt_rn(var + 1e-5f);
      for (int d = 0; d < 64; ++d) {
        const float y = ((float)hb[tid * 64 + d] - mean) * inv * ln2_s[l * 64 + d] + ln2_b[l * 64 + d];
        hb[tid * 64 + d] = (__bf16)y;
      }
    }
    __syncthreads();
  }

  // ---- head: logits = h @ head_w^T + head_b (head padded to 64 cols) ----
  for (int nt = 0; nt < 4; ++nt) {
    const int nb = nt * 16;
    v8f acc = {};
#pragma unroll
    for (int ks = 0; ks < 2; ++ks)
      acc = wmma_bf16(load_A(hb, 64, mbase, ks * 32), load_B(wHead, 64, nb, ks * 32), acc);
    const int n = nb + (lane & 15);
    if (n < 50) {
      const float bias = head_b[n];
      const int mb = mbase + ((lane >> 4) << 3);
#pragma unroll
      for (int r = 0; r < 8; ++r)
        out[((size_t)seq * 64 + mb + r) * 50 + n] = acc[r] + bias;
    }
  }
}

extern "C" void kernel_launch(void* const* d_in, const int* in_sizes, int n_in,
                              void* d_out, int out_size, void* d_ws, size_t ws_size,
                              hipStream_t stream) {
  const int*           x      = (const int*)d_in[0];
  const unsigned char* pm     = (const unsigned char*)d_in[1];
  const float*         emb    = (const float*)d_in[2];
  const float*         pos    = (const float*)d_in[3];
  const float*         in_w   = (const float*)d_in[4];
  const float*         in_b   = (const float*)d_in[5];
  const float*         out_w  = (const float*)d_in[6];
  const float*         out_b  = (const float*)d_in[7];
  const float*         ln1_s  = (const float*)d_in[8];
  const float*         ln1_b  = (const float*)d_in[9];
  const float*         ff1_w  = (const float*)d_in[10];
  const float*         ff1_b  = (const float*)d_in[11];
  const float*         ff2_w  = (const float*)d_in[12];
  const float*         ff2_b  = (const float*)d_in[13];
  const float*         ln2_s  = (const float*)d_in[14];
  const float*         ln2_b  = (const float*)d_in[15];
  const float*         head_w = (const float*)d_in[16];
  const float*         head_b = (const float*)d_in[17];

  __bf16* wbf = (__bf16*)d_ws;
  const int B = in_sizes[0] / 64;  // number of sequences

  cvtw<<<(WS_TOTAL + 255) / 256, 256, 0, stream>>>(in_w, out_w, ff1_w, ff2_w, head_w, wbf);
  tfwd<<<B, 128, 0, stream>>>(x, pm, emb, pos, in_b, out_b, ln1_s, ln1_b, ff1_b, ff2_b,
                              ln2_s, ln2_b, head_b,
                              wbf + WS_INW, wbf + WS_OUTW, wbf + WS_FF1, wbf + WS_FF2,
                              wbf + WS_HEAD, (float*)d_out);
}
